// LMoEWithConv_54546084659222
// MI455X (gfx1250) — matrix operand order repaired and checked by reading, not compiled, not run on hardware
//
#include <hip/hip_runtime.h>
#include <hip/hip_bf16.h>

#define B_  8
#define C_  512
#define H_  64
#define W_  64
#define E_  8
#define R_  64
#define TK_ 2

typedef __attribute__((ext_vector_type(16))) __bf16 v16bf;
typedef __attribute__((ext_vector_type(8)))  __bf16 v8bf;
typedef __attribute__((ext_vector_type(2)))  __bf16 v2bf;
typedef __attribute__((ext_vector_type(8)))  float  v8f;
typedef __attribute__((ext_vector_type(4)))  float  v4f;

__device__ __forceinline__ __bf16 to_bf(float f) { return (__bf16)f; }

// Low 32 bits of a flat shared pointer == LDS byte offset (ISA 10.2 aperture rule)
__device__ __forceinline__ unsigned lds_off(const void* p) {
  return (unsigned)(size_t)p;
}

// CDNA5 async DMA: global -> LDS, 16 bytes per lane, tracked by ASYNCcnt.
__device__ __forceinline__ void async_load_b128(unsigned lds_byte_off, const void* gaddr) {
  asm volatile("global_load_async_to_lds_b128 %0, %1, off"
               :: "v"(lds_byte_off), "v"(gaddr) : "memory");
}
__device__ __forceinline__ void wait_async0() {
  asm volatile("s_wait_asynccnt 0x0" ::: "memory");
}

// A fragment (bf16 16x32, row-major tile [row][rowstride] in LDS).
__device__ __forceinline__ v16bf load_fragA(const __bf16* tile, int rowstride, int koff) {
  const int lane = threadIdx.x & 31;
  const int row  = lane & 15;
  const int kb   = (lane < 16) ? 0 : 8;
  const __bf16* p = tile + row * rowstride + koff + kb;
  union { v16bf v; v8bf h[2]; } u;
  u.h[0] = *(const v8bf*)(p);
  u.h[1] = *(const v8bf*)(p + 16);
  return u.v;
}

// B fragment (bf16 32x16), tile stored transposed in LDS as [n][rowstride(K)].
__device__ __forceinline__ v16bf load_fragB(const __bf16* tile, int rowstride, int koff) {
  const int lane = threadIdx.x & 31;
  const int n    = lane & 15;
  const int kb   = (lane < 16) ? 0 : 8;
  const __bf16* p = tile + n * rowstride + koff + kb;
  union { v16bf v; v8bf h[2]; } u;
  u.h[0] = *(const v8bf*)(p);
  u.h[1] = *(const v8bf*)(p + 16);
  return u.v;
}

__device__ __forceinline__ v8f wmma_bf16(v16bf a, v16bf b, v8f c) {
  return __builtin_amdgcn_wmma_f32_16x16x32_bf16(false, a, false, b, (short)0, c, false, false);
}

// ---------------------------------------------------------------------------
// Kernel 1: xmean[b,c] = mean over HxW of x   (gating is linear -> mean first)
// ---------------------------------------------------------------------------
__global__ __launch_bounds__(256) void mean_kernel(const float* __restrict__ x,
                                                   float* __restrict__ xmean) {
  __shared__ float red[256];
  const int bc = blockIdx.x;
  const v4f* p = (const v4f*)(x + (size_t)bc * (H_ * W_));
  float s = 0.f;
  for (int i = threadIdx.x; i < (H_ * W_) / 4; i += 256) {
    v4f v = p[i];
    s += v[0] + v[1] + v[2] + v[3];
  }
  red[threadIdx.x] = s;
  __syncthreads();
  for (int off = 128; off > 0; off >>= 1) {
    if (threadIdx.x < off) red[threadIdx.x] += red[threadIdx.x + off];
    __syncthreads();
  }
  if (threadIdx.x == 0) xmean[bc] = red[0] * (1.0f / (H_ * W_));
}

// ---------------------------------------------------------------------------
// Kernel 2: logits -> softmax -> top-2
// ---------------------------------------------------------------------------
__global__ __launch_bounds__(64) void gate_kernel(const float* __restrict__ xmean,
                                                  const float* __restrict__ wG,
                                                  const float* __restrict__ bG,
                                                  int* __restrict__ tidx,
                                                  float* __restrict__ tg,
                                                  float* __restrict__ gw_out) {
  __shared__ float lg[B_ * E_];
  const int t = threadIdx.x;
  const int b = t >> 3, e = t & 7;
  float s = bG[e];
  for (int c = 0; c < C_; ++c) s += xmean[b * C_ + c] * wG[e * C_ + c];
  lg[t] = s;
  __syncthreads();
  if (e == 0) {
    float p[E_];
    float mx = lg[b * E_];
    for (int j = 1; j < E_; ++j) mx = fmaxf(mx, lg[b * E_ + j]);
    float den = 0.f;
    for (int j = 0; j < E_; ++j) { p[j] = __expf(lg[b * E_ + j] - mx); den += p[j]; }
    const float inv = 1.f / den;
    for (int j = 0; j < E_; ++j) p[j] *= inv;
    int i0 = 0;
    for (int j = 1; j < E_; ++j) if (p[j] > p[i0]) i0 = j;
    int i1 = (i0 == 0) ? 1 : 0;
    for (int j = 0; j < E_; ++j) if (j != i0 && p[j] > p[i1]) i1 = j;
    tidx[b * TK_ + 0] = i0;   tidx[b * TK_ + 1] = i1;
    tg[b * TK_ + 0]   = p[i0]; tg[b * TK_ + 1]  = p[i1];
    gw_out[b * TK_ + 0] = p[i0]; gw_out[b * TK_ + 1] = p[i1];
  }
}

// ---------------------------------------------------------------------------
// Prep kernels: one-time bf16 repacks in WMMA-friendly layouts.
// ---------------------------------------------------------------------------
// xT: NHWC bf16  xT[((b*H+y)*W+p)*C + c]
__global__ __launch_bounds__(256) void prep_x(const float* __restrict__ x,
                                              __bf16* __restrict__ xT) {
  __shared__ __bf16 t[64 * 32];                // [pixel][ch]
  const int blk = blockIdx.x;                  // (b*H + y)*16 + chunk
  const int cc  = (blk & 15) * 32;
  const int by  = blk >> 4;                    // b*H + y
  const int tid = threadIdx.x;
  {
    const int cl = (tid & 15) * 2;
    const int p0 = (tid >> 4) * 4;
    const float* xr0 = x + (((size_t)(by >> 6) * C_ + cc + cl) * H_ + (by & 63)) * W_;
    const float* xr1 = xr0 + (size_t)H_ * W_;
    #pragma unroll
    for (int i = 0; i < 4; ++i) {
      const v2bf pr = {to_bf(xr0[p0 + i]), to_bf(xr1[p0 + i])};
      *(v2bf*)&t[(p0 + i) * 32 + cl] = pr;
    }
  }
  __syncthreads();
  {
    const int px = tid >> 2, q = tid & 3;
    const v8bf v = *(const v8bf*)&t[px * 32 + q * 8];
    *(v8bf*)&xT[((size_t)by * W_ + px) * C_ + cc + q * 8] = v;
  }
}

// wAT: [e][tap(=ty*3+tx)][r][c] bf16
__global__ __launch_bounds__(256) void prep_wA(const float* __restrict__ wA,
                                               __bf16* __restrict__ wAT) {
  const int idx = blockIdx.x * 256 + threadIdx.x;   // (e,r,cpair): E*R*C/2 = 131072
  const int cp  = idx & (C_ / 2 - 1);
  const int er  = idx / (C_ / 2);                   // e*R + r
  const float* w0 = wA + ((size_t)er * C_ + cp * 2) * 9;
  const float* w1 = w0 + 9;
  const int e = er >> 6, r = er & 63;
  #pragma unroll
  for (int t = 0; t < 9; ++t) {
    const v2bf pr = {to_bf(w0[t]), to_bf(w1[t])};
    *(v2bf*)&wAT[(((size_t)e * 9 + t) * R_ + r) * C_ + cp * 2] = pr;
  }
}

// wBT: [e][c][r] bf16 (same layout as wB, just converted)
__global__ __launch_bounds__(256) void prep_wB(const float* __restrict__ wB,
                                               __bf16* __restrict__ wBT) {
  const int i = blockIdx.x * 256 + threadIdx.x;     // pairs: E*C*R/2 = 131072
  const v2bf pr = {to_bf(wB[i * 2]), to_bf(wB[i * 2 + 1])};
  *(v2bf*)&wBT[i * 2] = pr;
}

// ---------------------------------------------------------------------------
// Kernel 3: fused convA(3x3,512->64) -> convB(1x1,64->512) -> g*(y+bB)*x
// Staging is pure async DMA (global_load_async_to_lds_b128), double-buffered,
// software-pipelined under the WMMAs. OOB tap rows skipped entirely.
// ---------------------------------------------------------------------------
__global__ __launch_bounds__(256) void moe_main(const __bf16* __restrict__ xT,
                                                const __bf16* __restrict__ wAT,
                                                const __bf16* __restrict__ wBT,
                                                const float* __restrict__ x,
                                                const float* __restrict__ bA,
                                                const float* __restrict__ bB,
                                                const int* __restrict__ tidx,
                                                const float* __restrict__ tg,
                                                float* __restrict__ out) {
  __shared__ __bf16 ldsA[2][66 * 32];     // padded x tile: [pixel -1..64][ch 32]
  __shared__ __bf16 ldsBt[2][3][64 * 32]; // wA^T per tap:  [rank 64][ch 32]
  __shared__ __bf16 ldsH[64 * 64];        // h tile:        [pixel 64][rank 64]
  __shared__ __bf16 ldsW[128 * 64];       // wB chunk^T:    [c_out 128][rank 64]
  __shared__ float  ldsS[8 * 256];        // per-wave 16x16 f32 transpose scratch

  const int tid  = threadIdx.x;
  const int wv   = tid >> 5;
  const int lane = tid & 31;
  const int bb   = blockIdx.x >> 6;   // batch
  const int y    = blockIdx.x & 63;   // output row
  const int m    = wv & 3;            // M tile (16-pixel group)
  const int ng   = wv >> 2;           // N group (0/1)

  const unsigned offA  = lds_off(&ldsA[0][0]);
  const unsigned offBt = lds_off(&ldsBt[0][0][0]);
  const unsigned offW  = lds_off(&ldsW[0]);

  // zero the pad pixel-columns (rows 0 and 65) of both buffers, once
  if (tid < 64) {
    const int pb  = tid >> 5;
    const int row = ((tid >> 4) & 1) ? 65 : 0;
    const v2bf z = {to_bf(0.f), to_bf(0.f)};
    *(v2bf*)&ldsA[pb][row * 32 + (tid & 15) * 2] = z;
  }

  // valid 3x3 tap rows for this output row (uniform; OOB rows contribute 0)
  int tys[3]; int nty = 0;
  #pragma unroll
  for (int ty = 0; ty < 3; ++ty) {
    const int yy = y + ty - 1;
    if (yy >= 0 && yy < H_) tys[nty++] = ty;
  }
  const int total = nty * 16;           // chunks of 32 channels

  const int px = tid >> 2;              // async lane mapping: 0..63
  const int q  = tid & 3;               // 16B quarter within 64B row

  #pragma unroll
  for (int k = 0; k < TK_; ++k) {
    const int   e = tidx[bb * TK_ + k];
    const float g = tg[bb * TK_ + k];

    // issue one chunk's DMA: 1x A-tile row-slice + 3 tap B-tiles per thread
    auto issue_chunk = [&](int p, int ty, int cc) {
      const int yy = y + ty - 1;
      const __bf16* ga = xT + ((size_t)(bb * H_ + yy) * W_ + px) * C_ + cc + q * 8;
      async_load_b128(offA + (unsigned)p * (66 * 32 * 2) + (px + 1) * 64 + q * 16, ga);
      const __bf16* plane = wAT + ((size_t)(e * 9 + ty * 3) * R_ + px) * C_ + cc + q * 8;
      #pragma unroll
      for (int tx = 0; tx < 3; ++tx)
        async_load_b128(offBt + (unsigned)(p * 3 + tx) * (64 * 32 * 2) + px * 64 + q * 16,
                        plane + (size_t)tx * R_ * C_);
    };

    // ------------------ convA: software-pipelined K loop ------------------
    v8f accA0 = {}; v8f accA1 = {};
    issue_chunk(0, tys[0], 0);
    wait_async0();
    __syncthreads();
    for (int ci = 0; ci < total; ++ci) {
      const int p = ci & 1;
      if (ci + 1 < total)               // DMA next chunk under the WMMAs
        issue_chunk(1 - p, tys[(ci + 1) >> 4], ((ci + 1) & 15) * 32);
      const __bf16* A = &ldsA[p][0];
      #pragma unroll
      for (int tx = 0; tx < 3; ++tx) {
        const v16bf a  = load_fragA(A + (m * 16 + tx) * 32, 32, 0);
        const v16bf b0 = load_fragB(&ldsBt[p][tx][0] + (ng * 2 + 0) * 16 * 32, 32, 0);
        const v16bf b1 = load_fragB(&ldsBt[p][tx][0] + (ng * 2 + 1) * 16 * 32, 32, 0);
        accA0 = wmma_bf16(a, b0, accA0);
        accA1 = wmma_bf16(a, b1, accA1);
      }
      if (ci + 1 < total) {
        wait_async0();                  // next chunk landed
        __syncthreads();                // all compute(ci) done before reuse
      }
    }
    // h (+ bias bA) -> LDS. C/D layout: VGPR v -> M = v + 8*(lane>=16), N = lane&15.
    {
      const int ncol = lane & 15;
      #pragma unroll
      for (int v = 0; v < 8; ++v) {
        const int mr  = m * 16 + v + ((lane >> 4) * 8);
        const int n0c = (ng * 2 + 0) * 16 + ncol;
        const int n1c = (ng * 2 + 1) * 16 + ncol;
        ldsH[mr * 64 + n0c] = to_bf(accA0[v] + bA[e * R_ + n0c]);
        ldsH[mr * 64 + n1c] = to_bf(accA1[v] + bA[e * R_ + n1c]);
      }
    }
    __syncthreads();

    // ------------------ convB + fused coalesced epilogue ------------------
    const v16bf a0 = load_fragA(ldsH + m * 16 * 64, 64, 0);   // K = 0..31
    const v16bf a1 = load_fragA(ldsH + m * 16 * 64, 64, 32);  // K = 32..63
    float* sc = ldsS + wv * 256;                              // wave-private
    for (int nc = 0; nc < 4; ++nc) {            // 4 chunks of 128 c_out
      {                                         // DMA wB chunk: 64B per thread
        const __bf16* wsrc = wBT + ((size_t)e * C_ + nc * 128) * R_ + tid * 32;
        #pragma unroll
        for (int t = 0; t < 4; ++t)
          async_load_b128(offW + tid * 64 + t * 16, wsrc + t * 8);
      }
      wait_async0();
      __syncthreads();
      #pragma unroll
      for (int j = 0; j < 4; ++j) {
        const int nl = ng * 4 + j;              // n tile within chunk: 0..7
        const int cb = nc * 128 + nl * 16;      // c_out base
        const v16bf bf0 = load_fragB(ldsW + nl * 16 * 64, 64, 0);
        const v16bf bf1 = load_fragB(ldsW + nl * 16 * 64, 64, 32);
        v8f acc = {};
        acc = wmma_bf16(a0, bf0, acc);
        acc = wmma_bf16(a1, bf1, acc);
        // dump tile (bias added) to wave-private scratch, c-major [n][m]
        {
          const int  nn  = lane & 15;
          const float bBc = bB[e * C_ + cb + nn];
          v8f st;
          #pragma unroll
          for (int v = 0; v < 8; ++v) st[v] = acc[v] + bBc;
          *(v8f*)&sc[nn * 16 + (lane >> 4) * 8] = st;   // DS in-order per wave
        }
        // re-read transposed: lane -> (channel, 8 consecutive pixels)
        {
          const int cl2 = lane >> 1;
          const int ph  = (lane & 1) * 8;
          const v8f t = *(const v8f*)&sc[cl2 * 16 + ph];
          const int c = cb + cl2;
          const size_t base = (((size_t)bb * C_ + c) * H_ + y) * W_ + m * 16 + ph;
          #pragma unroll
          for (int i = 0; i < 8; ++i) {
            const float val = g * t[i] * x[base + i];
            if (k == 0) out[base + i] = val;
            else        out[base + i] += val;   // same WG owns the row: no race
          }
        }
      }
      __syncthreads();
    }
  }
}

extern "C" void kernel_launch(void* const* d_in, const int* in_sizes, int n_in,
                              void* d_out, int out_size, void* d_ws, size_t ws_size,
                              hipStream_t stream) {
  const float* x  = (const float*)d_in[0];
  const float* wA = (const float*)d_in[1];
  const float* bA = (const float*)d_in[2];
  const float* wB = (const float*)d_in[3];
  const float* bB = (const float*)d_in[4];
  const float* wG = (const float*)d_in[5];
  const float* bG = (const float*)d_in[6];
  float* out = (float*)d_out;

  // workspace layout (16B-aligned sections)
  char* ws = (char*)d_ws;
  float*  xmean = (float*)ws;                               ws += (size_t)B_ * C_ * 4;
  int*    tidx  = (int*)ws;                                 ws += (size_t)B_ * TK_ * 4;
  float*  tg    = (float*)ws;                               ws += (size_t)B_ * TK_ * 4;
  ws = (char*)(((size_t)ws + 255) & ~(size_t)255);
  __bf16* xT    = (__bf16*)ws;                              ws += (size_t)B_ * H_ * W_ * C_ * 2;
  __bf16* wAT   = (__bf16*)ws;                              ws += (size_t)E_ * 9 * R_ * C_ * 2;
  __bf16* wBT   = (__bf16*)ws;

  float* gw_out = out + (size_t)B_ * C_ * H_ * W_;          // tuple tail

  mean_kernel<<<B_ * C_, 256, 0, stream>>>(x, xmean);
  gate_kernel<<<1, 64, 0, stream>>>(xmean, wG, bG, tidx, tg, gw_out);
  prep_wB<<<(E_ * C_ * R_ / 2) / 256, 256, 0, stream>>>(wB, wBT);
  prep_wA<<<(E_ * R_ * C_ / 2) / 256, 256, 0, stream>>>(wA, wAT);
  prep_x<<<B_ * H_ * 16, 256, 0, stream>>>(x, xT);
  moe_main<<<B_ * H_, 256, 0, stream>>>(xT, wAT, wBT, x, bA, bB, tidx, tg, out);
}
